// MultiAttentionLayer_58428735095550
// MI455X (gfx1250) — matrix operand order repaired and checked by reading, not compiled, not run on hardware
//
#include <hip/hip_runtime.h>
#include <hip/hip_bf16.h>

// ---------------- problem constants ----------------
constexpr int N      = 4096;
constexpr int IN_DIM = 512;
constexpr int HID    = 256;
constexpr int HEADS  = 2;
constexpr int OUT    = 40;
constexpr int OUTP   = 48;            // OUT padded to 3x16 tiles
constexpr int TWOH   = HEADS * HID;   // 512
#define LN_EPS 1e-5f

// ---------------- types ----------------
typedef __attribute__((ext_vector_type(16))) __bf16        v16bf;
typedef __attribute__((ext_vector_type(8)))  float         v8f;
typedef __attribute__((ext_vector_type(8)))  unsigned int  v8u;

#define DEV __device__ __forceinline__

// ---------------- helpers ----------------
DEV unsigned short f2bf(float f) {
  unsigned int u = __builtin_bit_cast(unsigned int, f);
  unsigned int r = u + 0x7FFFu + ((u >> 16) & 1u);   // round-to-nearest-even
  return (unsigned short)(r >> 16);
}
DEV unsigned int pack2bf(float lo, float hi) {
  return (unsigned int)f2bf(lo) | ((unsigned int)f2bf(hi) << 16);
}

DEV v8u mk8(uint4 x, uint4 y) {
  v8u r;
  r[0] = x.x; r[1] = x.y; r[2] = x.z; r[3] = x.w;
  r[4] = y.x; r[5] = y.y; r[6] = y.z; r[7] = y.w;
  return r;
}
// A-matrix (16x32 bf16) operand: words [wbase+half*4 .. +3] and [wbase+8+half*4 .. +3]
DEV v8u load_a(const unsigned int* p, int wbase, int half) {
  uint4 x = *(const uint4*)(p + wbase + half * 4);
  uint4 y = *(const uint4*)(p + wbase + 8 + half * 4);
  return mk8(x, y);
}
// B-matrix (32x16 bf16) operand: words [wbase+half*4 .. +3] and [wbase+8+half*4 .. +3]
DEV v8u load_b(const unsigned int* p, int wbase, int half) {
  uint4 x = *(const uint4*)(p + wbase + half * 4);
  uint4 y = *(const uint4*)(p + wbase + 8 + half * 4);
  return mk8(x, y);
}

DEV v8f vzero() {
  v8f z;
#pragma unroll
  for (int i = 0; i < 8; ++i) z[i] = 0.f;
  return z;
}

DEV v8f wmma_bf16(v8u a, v8u b, v8f c) {
  return __builtin_amdgcn_wmma_f32_16x16x32_bf16(
      false, __builtin_bit_cast(v16bf, a),
      false, __builtin_bit_cast(v16bf, b),
      (short)0, c, false, false);
}
// reductions across the 16-lane half holding one C-layout row
DEV float hmax16(float x) {
#pragma unroll
  for (int m = 1; m < 16; m <<= 1) x = fmaxf(x, __shfl_xor(x, m, 32));
  return x;
}
DEV float hsum16(float x) {
#pragma unroll
  for (int m = 1; m < 16; m <<= 1) x += __shfl_xor(x, m, 32);
  return x;
}

// ---------------- K0: bf16 convert + weight transposes ----------------
__global__ void prep_kernel(const float* __restrict__ h,
                            const float* __restrict__ Wq, const float* __restrict__ Wk,
                            const float* __restrict__ Wv, const float* __restrict__ Wo,
                            const float* __restrict__ Wp,
                            unsigned short* hb, unsigned short* WqT, unsigned short* WkT,
                            unsigned short* WvT, unsigned short* WoT, unsigned short* WpT) {
  int tid = blockIdx.x * blockDim.x + threadIdx.x;
  int stride = gridDim.x * blockDim.x;
  for (int i = tid; i < N * IN_DIM; i += stride) hb[i] = f2bf(h[i]);
  for (int i = tid; i < HEADS * IN_DIM * HID; i += stride) {
    int hd = i / (IN_DIM * HID);
    int r  = i % (IN_DIM * HID);
    int k  = r / HID, d = r % HID;                  // W[hd][k][d]
    int o  = (hd * HID + d) * IN_DIM + k;           // WT[hd][d][k]
    WqT[o] = f2bf(Wq[i]); WkT[o] = f2bf(Wk[i]); WvT[o] = f2bf(Wv[i]);
  }
  for (int i = tid; i < TWOH * IN_DIM; i += stride) {   // Wo[k][n] -> WoT[n][k]
    int k = i / IN_DIM, n = i % IN_DIM;
    WoT[n * TWOH + k] = f2bf(Wo[i]);
  }
  for (int i = tid; i < OUTP * IN_DIM; i += stride) {   // Wp[k][n] -> WpT[n][k], zero-pad n>=40
    int n = i / IN_DIM, k = i % IN_DIM;
    WpT[n * IN_DIM + k] = (n < OUT) ? f2bf(Wp[k * OUT + n]) : (unsigned short)0;
  }
}

// ---------------- K1: Q,K,V projections (bf16 WMMA) ----------------
__global__ void qkv_kernel(const unsigned short* __restrict__ hb,
                           const unsigned short* __restrict__ WqT,
                           const unsigned short* __restrict__ WkT,
                           const unsigned short* __restrict__ WvT,
                           const float* __restrict__ bq, const float* __restrict__ bk,
                           const float* __restrict__ bv,
                           unsigned short* Qb, unsigned short* Kb, unsigned short* Vt) {
  int wave = threadIdx.x >> 5, lane = threadIdx.x & 31;
  int g = blockIdx.x * (blockDim.x >> 5) + wave;              // 0..8191
  int hd = g >> 12, r = g & 4095, mt = r >> 4, dt = r & 15;
  int half = lane >> 4, l16 = lane & 15;

  const unsigned int* hw = (const unsigned int*)hb;
  const unsigned int* qw = (const unsigned int*)WqT;
  const unsigned int* kw = (const unsigned int*)WkT;
  const unsigned int* vw = (const unsigned int*)WvT;

  v8f cq = vzero(), ck = vzero(), cv = vzero();
  int aw = ((mt * 16 + l16) * IN_DIM) >> 1;                  // word base, A rows
  int bw = ((hd * HID + dt * 16 + l16) * IN_DIM) >> 1;       // word base, B rows
#pragma unroll
  for (int s = 0; s < IN_DIM / 32; ++s) {
    v8u a  = load_a(hw, aw + s * 16, half);
    v8u b0 = load_b(qw, bw + s * 16, half);
    v8u b1 = load_b(kw, bw + s * 16, half);
    v8u b2 = load_b(vw, bw + s * 16, half);
    cq = wmma_bf16(a, b0, cq);
    ck = wmma_bf16(a, b1, ck);
    cv = wmma_bf16(a, b2, cv);
  }
  int col = dt * 16 + l16;
  float bqv = bq[hd * HID + col], bkv = bk[hd * HID + col], bvv = bv[hd * HID + col];
#pragma unroll
  for (int rr = 0; rr < 8; ++rr) {
    int row = mt * 16 + rr + 8 * half;
    Qb[(hd * N + row) * HID + col] = f2bf(cq[rr] + bqv);
    Kb[(hd * N + row) * HID + col] = f2bf(ck[rr] + bkv);
    Vt[(hd * HID + col) * N + row] = f2bf(cv[rr] + bvv); // transposed for PV B-operand
  }
}

// ---------------- K2: fused masked attention (flash-style, split-K in block) ----
// 512 blocks (one per head x 16-row tile) x 4 waves; each wave owns an
// interleaved quarter of the 4096 keys, block combines (m, l, X) in LDS.
#define ATT_WAVES 4
__global__ void attn_kernel(const unsigned short* __restrict__ Qb,
                            const unsigned short* __restrict__ Kb,
                            const unsigned short* __restrict__ Vt,
                            const float* __restrict__ adj,
                            unsigned short* Xb) {
  __shared__ uint4 qlds[32][16];                     // shared Q tile, A-operand order (8 KB)
  __shared__ float pst[ATT_WAVES][16 * 32];          // per-wave P relayout (8 KB)
  __shared__ float xacc[16][256];                    // block X combine (16 KB)
  __shared__ float mbuf[ATT_WAVES][16];              // per-wave row maxima
  __shared__ float lbuf[ATT_WAVES][16];              // per-wave row sums
  int wave = threadIdx.x >> 5, lane = threadIdx.x & 31;
  int g = blockIdx.x;                                // 0..511
  int hd = g >> 8, mt = g & 255;
  int half = lane >> 4, l16 = lane & 15;

  const unsigned int* Qw = (const unsigned int*)Qb;
  const unsigned int* Kw = (const unsigned int*)Kb;
  const unsigned int* Vw = (const unsigned int*)Vt;

  // zero the combine buffer
  for (int i = threadIdx.x; i < 16 * 256; i += 32 * ATT_WAVES) (&xacc[0][0])[i] = 0.f;
  // wave 0 stages the Q tile (16 rows x 256) as pre-swizzled A-operand uint4 pairs
  if (wave == 0) {
    int qw0 = ((hd * N + mt * 16 + l16) * HID) >> 1;
#pragma unroll
    for (int s = 0; s < 8; ++s) {
      qlds[lane][s * 2]     = *(const uint4*)(Qw + qw0 + s * 16 + half * 4);
      qlds[lane][s * 2 + 1] = *(const uint4*)(Qw + qw0 + s * 16 + 8 + half * 4);
    }
  }
  __syncthreads();

  float mrun[8], lrun[8];
  v8f X[16];
#pragma unroll
  for (int r = 0; r < 8; ++r) { mrun[r] = -__builtin_inff(); lrun[r] = 0.f; }
#pragma unroll
  for (int t = 0; t < 16; ++t) X[t] = vzero();

  for (int jb = wave; jb < N / 32; jb += ATT_WAVES) {
    int c0 = jb * 32;
    // ---- S = Q K^T for two 16-col tiles ----
    v8f S0 = vzero(), S1 = vzero();
    int n0 = ((hd * N + c0 + l16) * HID) >> 1;
    int n1 = n0 + 8 * HID;
#pragma unroll
    for (int s = 0; s < 8; ++s) {
      v8u a  = mk8(qlds[lane][s * 2], qlds[lane][s * 2 + 1]);
      v8u b0 = load_b(Kw, n0 + s * 16, half);
      v8u b1 = load_b(Kw, n1 + s * 16, half);
      S0 = wmma_bf16(a, b0, S0);
      S1 = wmma_bf16(a, b1, S1);
    }
    // ---- adj mask-multiply + online softmax ----
#pragma unroll
    for (int r = 0; r < 8; ++r) {
      int row = mt * 16 + r + 8 * half;
      size_t ab = ((size_t)hd * N + row) * N + c0;
      float a0 = __builtin_nontemporal_load(adj + ab + l16);        // adj streamed once
      float a1 = __builtin_nontemporal_load(adj + ab + 16 + l16);
      float s0 = S0[r] * a0;
      float s1 = S1[r] * a1;
      float tm = hmax16(fmaxf(s0, s1));
      float mn = fmaxf(mrun[r], tm);
      float alpha = __expf(mrun[r] - mn);
      mrun[r] = mn;
      float p0 = __expf(s0 - mn), p1 = __expf(s1 - mn);
      lrun[r] = lrun[r] * alpha + hsum16(p0 + p1);
      pst[wave][(r + 8 * half) * 32 + l16]      = p0;
      pst[wave][(r + 8 * half) * 32 + 16 + l16] = p1;
#pragma unroll
      for (int t = 0; t < 16; ++t) X[t][r] *= alpha;   // rescale accumulators
    }
    // ---- relayout P (C-layout f32 -> A-layout bf16) via per-wave LDS ----
    const float4* pr0 = (const float4*)&pst[wave][l16 * 32 + half * 8];
    const float4* pr1 = (const float4*)&pst[wave][l16 * 32 + 16 + half * 8];
    float4 f0 = pr0[0], f1 = pr0[1], f2 = pr1[0], f3 = pr1[1];
    v8u pa;
    pa[0] = pack2bf(f0.x, f0.y); pa[1] = pack2bf(f0.z, f0.w);
    pa[2] = pack2bf(f1.x, f1.y); pa[3] = pack2bf(f1.z, f1.w);
    pa[4] = pack2bf(f2.x, f2.y); pa[5] = pack2bf(f2.z, f2.w);
    pa[6] = pack2bf(f3.x, f3.y); pa[7] = pack2bf(f3.z, f3.w);
    // ---- X += P V : 16 output tiles over d = 0..255 ----
#pragma unroll
    for (int t = 0; t < 16; ++t) {
      int vb = (((hd * HID + t * 16 + l16) * N) >> 1) + (c0 >> 1);
      v8u b = load_b(Vw, vb, half);
      X[t] = wmma_bf16(pa, b, X[t]);
    }
  }

  // ---- publish per-wave softmax stats ----
  if (l16 == 0) {
#pragma unroll
    for (int r = 0; r < 8; ++r) {
      mbuf[wave][r + 8 * half] = mrun[r];
      lbuf[wave][r + 8 * half] = lrun[r];
    }
  }
  __syncthreads();
  // ---- global row max and per-wave rescale factor ----
  float scale[8];
#pragma unroll
  for (int r = 0; r < 8; ++r) {
    int row = r + 8 * half;
    float mg = -__builtin_inff();
#pragma unroll
    for (int w = 0; w < ATT_WAVES; ++w) mg = fmaxf(mg, mbuf[w][row]);
    scale[r] = __expf(mrun[r] - mg);
  }
  // ---- barrier-serialized (deterministic) combine into xacc ----
  for (int w = 0; w < ATT_WAVES; ++w) {
    if (wave == w) {
#pragma unroll
      for (int t = 0; t < 16; ++t)
#pragma unroll
        for (int r = 0; r < 8; ++r)
          xacc[r + 8 * half][t * 16 + l16] += X[t][r] * scale[r];
    }
    __syncthreads();
  }
  // ---- normalize and store: each wave stores 4 of the 16 d-tiles ----
#pragma unroll
  for (int r = 0; r < 8; ++r) {
    int row = r + 8 * half;
    float mg = -__builtin_inff();
#pragma unroll
    for (int w = 0; w < ATT_WAVES; ++w) mg = fmaxf(mg, mbuf[w][row]);
    float lg = 0.f;
#pragma unroll
    for (int w = 0; w < ATT_WAVES; ++w) lg += lbuf[w][row] * __expf(mbuf[w][row] - mg);
    float inv = 1.f / lg;
#pragma unroll
    for (int tt = 0; tt < 4; ++tt) {
      int t = wave * 4 + tt;
      int col = hd * HID + t * 16 + l16;
      Xb[(mt * 16 + row) * TWOH + col] = f2bf(xacc[row][t * 16 + l16] * inv);
    }
  }
}

// ---------------- K3a: z = Xb @ Wo + bo ----------------
__global__ void outproj_kernel(const unsigned short* __restrict__ Xb,
                               const unsigned short* __restrict__ WoT,
                               const float* __restrict__ bo, float* z) {
  int wave = threadIdx.x >> 5, lane = threadIdx.x & 31;
  int g = blockIdx.x * (blockDim.x >> 5) + wave;    // 0..8191
  int mt = g >> 5, nt = g & 31;
  int half = lane >> 4, l16 = lane & 15;
  const unsigned int* Aw = (const unsigned int*)Xb;
  const unsigned int* Bw = (const unsigned int*)WoT;
  v8f c = vzero();
  int aw = ((mt * 16 + l16) * TWOH) >> 1;
  int bw = ((nt * 16 + l16) * TWOH) >> 1;
#pragma unroll
  for (int s = 0; s < TWOH / 32; ++s) {
    v8u a = load_a(Aw, aw + s * 16, half);
    v8u b = load_b(Bw, bw + s * 16, half);
    c = wmma_bf16(a, b, c);
  }
  float bov = bo[nt * 16 + l16];
#pragma unroll
  for (int r = 0; r < 8; ++r) {
    int row = mt * 16 + r + 8 * half, col = nt * 16 + l16;
    z[row * IN_DIM + col] = c[r] + bov;
  }
}

// ---------------- K3b: layernorm -> bf16 M ----------------
__global__ void ln_kernel(const float* __restrict__ z, const float* __restrict__ gamma,
                          const float* __restrict__ beta, unsigned short* Mb) {
  __shared__ float red[256];
  int row = blockIdx.x, tid = threadIdx.x;
  float s = 0.f;
  for (int i = tid; i < IN_DIM; i += 256) s += z[row * IN_DIM + i];
  red[tid] = s; __syncthreads();
  for (int o = 128; o > 0; o >>= 1) { if (tid < o) red[tid] += red[tid + o]; __syncthreads(); }
  float mu = red[0] / IN_DIM; __syncthreads();
  float v = 0.f;
  for (int i = tid; i < IN_DIM; i += 256) { float d = z[row * IN_DIM + i] - mu; v += d * d; }
  red[tid] = v; __syncthreads();
  for (int o = 128; o > 0; o >>= 1) { if (tid < o) red[tid] += red[tid + o]; __syncthreads(); }
  float rstd = rsqrtf(red[0] / IN_DIM + LN_EPS);
  for (int i = tid; i < IN_DIM; i += 256)
    Mb[row * IN_DIM + i] = f2bf((z[row * IN_DIM + i] - mu) * rstd * gamma[i] + beta[i]);
}

// ---------------- K4: out = softmax(M @ Wp + bp) ----------------
__global__ void final_kernel(const unsigned short* __restrict__ Mb,
                             const unsigned short* __restrict__ WpT,
                             const float* __restrict__ bp, float* out) {
  int wave = threadIdx.x >> 5, lane = threadIdx.x & 31;
  int mt = blockIdx.x * (blockDim.x >> 5) + wave;    // 0..255
  int half = lane >> 4, l16 = lane & 15;
  const unsigned int* Aw = (const unsigned int*)Mb;
  const unsigned int* Bw = (const unsigned int*)WpT;
  v8f c[3];
#pragma unroll
  for (int t = 0; t < 3; ++t) c[t] = vzero();
  int aw = ((mt * 16 + l16) * IN_DIM) >> 1;
#pragma unroll
  for (int s = 0; s < IN_DIM / 32; ++s) {
    v8u a = load_a(Aw, aw + s * 16, half);
#pragma unroll
    for (int t = 0; t < 3; ++t) {
      int bw = (((t * 16 + l16) * IN_DIM) >> 1) + s * 16;
      v8u b = load_b(Bw, bw, half);
      c[t] = wmma_bf16(a, b, c[t]);
    }
  }
#pragma unroll
  for (int r = 0; r < 8; ++r) {
    float x[3], mx = -__builtin_inff();
#pragma unroll
    for (int t = 0; t < 3; ++t) {
      int col = t * 16 + l16;
      x[t] = (col < OUT) ? (c[t][r] + bp[col]) : -__builtin_inff();
      mx = fmaxf(mx, x[t]);
    }
    mx = hmax16(mx);
    float sum = 0.f;
#pragma unroll
    for (int t = 0; t < 3; ++t) {
      int col = t * 16 + l16;
      x[t] = (col < OUT) ? __expf(x[t] - mx) : 0.f;
      sum += x[t];
    }
    sum = hsum16(sum);
    int row = mt * 16 + r + 8 * half;
#pragma unroll
    for (int t = 0; t < 3; ++t) {
      int col = t * 16 + l16;
      if (col < OUT) out[row * OUT + col] = x[t] / sum;
    }
  }
}

// ---------------- launch ----------------
extern "C" void kernel_launch(void* const* d_in, const int* in_sizes, int n_in,
                              void* d_out, int out_size, void* d_ws, size_t ws_size,
                              hipStream_t stream) {
  const float* adj   = (const float*)d_in[0];
  const float* h     = (const float*)d_in[1];
  const float* Wq    = (const float*)d_in[2];
  const float* bq    = (const float*)d_in[3];
  const float* Wk    = (const float*)d_in[4];
  const float* bk    = (const float*)d_in[5];
  const float* Wv    = (const float*)d_in[6];
  const float* bv    = (const float*)d_in[7];
  const float* Wo    = (const float*)d_in[8];
  const float* bo    = (const float*)d_in[9];
  const float* gamma = (const float*)d_in[10];
  const float* beta  = (const float*)d_in[11];
  const float* Wp    = (const float*)d_in[12];
  const float* bp    = (const float*)d_in[13];
  float* out = (float*)d_out;

  char* ws = (char*)d_ws;
  size_t off = 0;
  auto alloc = [&](size_t bytes) { char* p = ws + off; off = (off + bytes + 255) & ~size_t(255); return p; };
  unsigned short* hb  = (unsigned short*)alloc((size_t)N * IN_DIM * 2);
  unsigned short* WqT = (unsigned short*)alloc((size_t)HEADS * HID * IN_DIM * 2);
  unsigned short* WkT = (unsigned short*)alloc((size_t)HEADS * HID * IN_DIM * 2);
  unsigned short* WvT = (unsigned short*)alloc((size_t)HEADS * HID * IN_DIM * 2);
  unsigned short* WoT = (unsigned short*)alloc((size_t)IN_DIM * TWOH * 2);
  unsigned short* WpT = (unsigned short*)alloc((size_t)OUTP * IN_DIM * 2);
  unsigned short* Qb  = (unsigned short*)alloc((size_t)HEADS * N * HID * 2);
  unsigned short* Kb  = (unsigned short*)alloc((size_t)HEADS * N * HID * 2);
  unsigned short* Vt  = (unsigned short*)alloc((size_t)HEADS * HID * N * 2);
  unsigned short* Xb  = (unsigned short*)alloc((size_t)N * TWOH * 2);
  float*          z   = (float*)alloc((size_t)N * IN_DIM * 4);
  unsigned short* Mb  = (unsigned short*)alloc((size_t)N * IN_DIM * 2);
  (void)ws_size; (void)in_sizes; (void)n_in; (void)out_size;

  prep_kernel<<<4096, 256, 0, stream>>>(h, Wq, Wk, Wv, Wo, Wp, hb, WqT, WkT, WvT, WoT, WpT);
  qkv_kernel<<<1024, 256, 0, stream>>>(hb, WqT, WkT, WvT, bq, bk, bv, Qb, Kb, Vt);
  attn_kernel<<<HEADS * (N / 16), 32 * ATT_WAVES, 0, stream>>>(Qb, Kb, Vt, adj, Xb);
  outproj_kernel<<<1024, 256, 0, stream>>>(Xb, WoT, bo, z);
  ln_kernel<<<N, 256, 0, stream>>>(z, gamma, beta, Mb);
  final_kernel<<<32, 256, 0, stream>>>(Mb, WpT, bp, out);
}
